// SelectiveSSMLayer_1090921693680
// MI455X (gfx1250) — matrix operand order repaired and checked by reading, not compiled, not run on hardware
//
#include <hip/hip_runtime.h>
#include <hip/hip_bf16.h>

#define D_MODEL 768
#define STATE_N 16
#define BATCH   8
#define SEQ     2048
#define MROWS   (BATCH * SEQ)   // 16384

typedef __attribute__((ext_vector_type(16))) __bf16 v16bf;
typedef __attribute__((ext_vector_type(8)))  __bf16 v8bf;
typedef __attribute__((ext_vector_type(8)))  float  v8f;

static __device__ __forceinline__ __bf16 f2bf(float f) {
  union { float f; unsigned u; } a; a.f = f;
  unsigned r = (a.u + 0x7FFFu + ((a.u >> 16) & 1u)) >> 16;   // round-to-nearest-even
  union { unsigned short s; __bf16 b; } o; o.s = (unsigned short)r;
  return o.b;
}

// ---------------- f32 -> bf16 convert ----------------
__global__ void cvt_f32_bf16(const float* __restrict__ src, __bf16* __restrict__ dst, int n) {
  int i = blockIdx.x * blockDim.x + threadIdx.x;
  if (i < n) dst[i] = f2bf(src[i]);
}

// ---------------- WMMA bf16 GEMM: out[M,N] = A[M,K] * W[N,K]^T (+bias, epilogue) ----
// K is compile-time (= D_MODEL): the k-loop is fully unrolled so every fragment load
// is "base + constant" and folds into the 24-bit instruction offset. Only two base
// address pairs stay live; the scheduler pipelines loads ahead of WMMAs within the
// launch_bounds-granted register budget (no manual ping-pong, no spills).
// Macro tile 64(M) x 32(N) per wave = 4x2 WMMA tiles: 12 b128 loads per 8 wmma.
// mode 0: store acc+bias ; mode 1: softplus(acc+bias) ; mode 2: store acc raw
// split>0: columns >= split go to out1 (col-split), both halves use ld = split
__global__ __launch_bounds__(256, 2)
void wmma_gemm_bf16(const __bf16* __restrict__ A,
                    const __bf16* __restrict__ W,
                    const float* __restrict__ bias,
                    float* __restrict__ out0, float* __restrict__ out1,
                    int split, int M, int N, int mode) {
  const int K = D_MODEL;                        // compile-time K
  int wave = (blockIdx.x * blockDim.x + threadIdx.x) >> 5;
  int lane = threadIdx.x & 31;
  int ntiles = N >> 5;                          // macro tiles are 32 wide
  int tiles  = (M >> 6) * ntiles;               // and 64 tall
  if (wave >= tiles) return;                    // whole-wave uniform: EXEC stays all-1s
  int tm = wave / ntiles, tn = wave % ntiles;
  int m0 = tm << 6, n0 = tn << 5;
  int half = lane >> 4;                         // 0: lanes 0-15, 1: lanes 16-31
  int lr   = lane & 15;

  // A fragment rows: lane half 0 -> K 0..15; half 1 -> K 16..31
  const __bf16* ar = A + (size_t)(m0 + lr) * K + half * 16;
  // B fragment: lane holds column; half 0 -> K 0..15, half 1 -> K 16..31 (contiguous 32B)
  const __bf16* br = W + (size_t)(n0 + lr) * K + half * 16;

  v8f c[4][2];
  #pragma unroll
  for (int mi = 0; mi < 4; ++mi)
    #pragma unroll
    for (int ni = 0; ni < 2; ++ni) c[mi][ni] = {};

  union Fr { v16bf v; v8bf h[2]; };

  #pragma unroll
  for (int k0 = 0; k0 < K; k0 += 32) {          // fully unrolled: 24 steps
    Fr a[4], b[2];
    #pragma unroll
    for (int mi = 0; mi < 4; ++mi) {
      // offset = (mi*16*K + k0) elements -> constant byte immediate per load
      a[mi].h[0] = *(const v8bf*)(ar + mi * (16 * K) + k0);
      a[mi].h[1] = *(const v8bf*)(ar + mi * (16 * K) + k0 + 8);
    }
    #pragma unroll
    for (int ni = 0; ni < 2; ++ni)
      b[ni].v = *(const v16bf*)(br + ni * (16 * K) + k0);
    #pragma unroll
    for (int mi = 0; mi < 4; ++mi)
      #pragma unroll
      for (int ni = 0; ni < 2; ++ni)
        c[mi][ni] = __builtin_amdgcn_wmma_f32_16x16x32_bf16(
            false, a[mi].v, false, b[ni].v, (short)0, c[mi][ni], false, false);
  }

  int ldo = split ? split : N;
  #pragma unroll
  for (int ni = 0; ni < 2; ++ni) {
    int col = n0 + 16 * ni + lr;
    float bval = (bias != nullptr) ? bias[col] : 0.0f;
    float* outp = out0;
    int ocol = col;
    if (split && col >= split) { outp = out1; ocol = col - split; }
    #pragma unroll
    for (int mi = 0; mi < 4; ++mi) {
      int rbase = m0 + 16 * mi + half * 8;      // C layout: VGPR i -> M=i (+8 hi half)
      #pragma unroll
      for (int i = 0; i < 8; ++i) {
        float v = c[mi][ni][i] + bval;
        if (mode == 1) v = (v > 20.f) ? v : __logf(1.f + __expf(v));  // softplus
        outp[(size_t)(rbase + i) * ldo + ocol] = v;
      }
    }
  }
}

// ---------------- depthwise conv (k=3, pad=1) + SiLU; write f32 + bf16 ----------------
__global__ void conv_silu(const float* __restrict__ x_in, const float* __restrict__ cw,
                          const float* __restrict__ cb,
                          float* __restrict__ xc, __bf16* __restrict__ xcb, int total) {
  int i = blockIdx.x * blockDim.x + threadIdx.x;
  if (i >= total) return;
  int d = i % D_MODEL;
  int m = i / D_MODEL;
  int l = m % SEQ;
  float w0 = cw[d * 3 + 0], w1 = cw[d * 3 + 1], w2 = cw[d * 3 + 2];
  float acc = x_in[i] * w1 + cb[d];
  if (l > 0)       acc += x_in[i - D_MODEL] * w0;
  if (l < SEQ - 1) acc += x_in[i + D_MODEL] * w2;
  float s = acc / (1.f + __expf(-acc));         // silu
  xc[i]  = s;
  xcb[i] = f2bf(s);
}

// ---------------- sequential selective scan ----------------
// thread = (b, d, n); 16 n-lanes reduce y via shfl_xor (wave32, width 16)
__global__ void ssm_scan(const float* __restrict__ delta, const float* __restrict__ bc,
                         const float* __restrict__ xc, const float* __restrict__ A_log,
                         float* __restrict__ y) {
  int b = blockIdx.y;
  int d = blockIdx.x * 16 + (threadIdx.x >> 4);
  int n = threadIdx.x & 15;
  float A = -__expf(A_log[d * STATE_N + n]);
  float h = 0.f;
  size_t baseD  = (size_t)b * SEQ * D_MODEL + d;
  size_t baseBC = (size_t)b * SEQ * 32;
  const float* dp = delta + baseD;
  const float* xp = xc + baseD;
  const float* bp = bc + baseBC + n;
  const float* cp = bc + baseBC + 16 + n;
  float* yp = y + baseD;
  for (int l = 0; l < SEQ; ++l) {
    // prefetch 16 serial steps ahead (global_prefetch_b8) to hide stream latency
    __builtin_prefetch(dp + 16 * D_MODEL, 0, 1);
    __builtin_prefetch(xp + 16 * D_MODEL, 0, 1);
    __builtin_prefetch(bp + 16 * 32, 0, 1);
    float dt = *dp;
    float xt = *xp;
    float bt = *bp;
    float ct = *cp;
    h = __expf(dt * A) * h + (dt * bt) * xt;
    float s = h * ct;
    s += __shfl_xor(s, 1, 16);
    s += __shfl_xor(s, 2, 16);
    s += __shfl_xor(s, 4, 16);
    s += __shfl_xor(s, 8, 16);
    if (n == 0) *yp = s;
    dp += D_MODEL; xp += D_MODEL; yp += D_MODEL;
    bp += 32; cp += 32;
  }
}

// ---------------- gate: (y + xc*D) * silu(z) -> bf16 ----------------
__global__ void gate_mul(const float* __restrict__ y, const float* __restrict__ xc,
                         const float* __restrict__ z, const float* __restrict__ Dp,
                         __bf16* __restrict__ ybf, int total) {
  int i = blockIdx.x * blockDim.x + threadIdx.x;
  if (i >= total) return;
  int d = i % D_MODEL;
  float zv = z[i];
  float g = zv / (1.f + __expf(-zv));
  ybf[i] = f2bf((y[i] + xc[i] * Dp[d]) * g);
}

extern "C" void kernel_launch(void* const* d_in, const int* in_sizes, int n_in,
                              void* d_out, int out_size, void* d_ws, size_t ws_size,
                              hipStream_t stream) {
  (void)in_sizes; (void)n_in; (void)out_size; (void)ws_size;
  const float* x     = (const float*)d_in[0];
  const float* A_log = (const float*)d_in[1];
  const float* D_p   = (const float*)d_in[2];
  const float* Bpw   = (const float*)d_in[3];
  const float* Cpw   = (const float*)d_in[4];
  const float* dw    = (const float*)d_in[5];
  const float* db    = (const float*)d_in[6];
  const float* inw   = (const float*)d_in[7];
  const float* inb   = (const float*)d_in[8];
  const float* outw  = (const float*)d_in[9];
  const float* outb  = (const float*)d_in[10];
  const float* convw = (const float*)d_in[11];
  const float* convb = (const float*)d_in[12];
  float* out = (float*)d_out;

  char* ws = (char*)d_ws;
  size_t off = 0;
  auto alloc = [&](size_t bytes) {
    char* p = ws + off;
    off += (bytes + 255) & ~(size_t)255;
    return p;
  };
  const size_t MD = (size_t)MROWS * D_MODEL;            // 12.58M elems
  __bf16* w_in_bf    = (__bf16*)alloc((size_t)2 * D_MODEL * D_MODEL * 2);
  __bf16* w_delta_bf = (__bf16*)alloc((size_t)D_MODEL * D_MODEL * 2);
  __bf16* w_bc_bf    = (__bf16*)alloc((size_t)32 * D_MODEL * 2);
  __bf16* w_out_bf   = (__bf16*)alloc((size_t)D_MODEL * D_MODEL * 2);
  __bf16* x_bf       = (__bf16*)alloc(MD * 2);          // reused as y_bf after GEMM1
  __bf16* xc_bf      = (__bf16*)alloc(MD * 2);
  float*  x_in       = (float*)alloc(MD * 4);           // reused as y after conv
  float*  z          = (float*)alloc(MD * 4);
  float*  xc         = (float*)alloc(MD * 4);
  float*  delta      = (float*)alloc(MD * 4);
  float*  bc         = (float*)alloc((size_t)MROWS * 32 * 4);
  float*  y   = x_in;            // x_in dead after conv
  __bf16* ybf = x_bf;            // x_bf dead after GEMM1

  auto cvt = [&](const float* s, __bf16* d, int n) {
    cvt_f32_bf16<<<(n + 255) / 256, 256, 0, stream>>>(s, d, n);
  };
  cvt(inw,  w_in_bf,    2 * D_MODEL * D_MODEL);
  cvt(dw,   w_delta_bf, D_MODEL * D_MODEL);
  cvt(outw, w_out_bf,   D_MODEL * D_MODEL);
  cvt(Bpw,  w_bc_bf,               STATE_N * D_MODEL);
  cvt(Cpw,  w_bc_bf + STATE_N * D_MODEL, STATE_N * D_MODEL);
  cvt(x, x_bf, (int)MD);

  auto gemm = [&](const __bf16* A, const __bf16* W, const float* bias,
                  float* o0, float* o1, int split, int N, int mode) {
    int waves  = (MROWS / 64) * (N / 32);       // 64x32 macro tiles
    int blocks = (waves * 32 + 255) / 256;
    wmma_gemm_bf16<<<blocks, 256, 0, stream>>>(A, W, bias, o0, o1, split,
                                               MROWS, N, mode);
  };

  // 1) xz = x @ in_w^T + in_b  -> x_in | z
  gemm(x_bf, w_in_bf, inb, x_in, z, D_MODEL, 2 * D_MODEL, 0);
  // 2) depthwise conv + silu
  conv_silu<<<((int)MD + 255) / 256, 256, 0, stream>>>(x_in, convw, convb, xc, xc_bf, (int)MD);
  // 3) delta = softplus(xc @ delta_w^T + delta_b)
  gemm(xc_bf, w_delta_bf, db, delta, nullptr, 0, D_MODEL, 1);
  // 4) [B_sel | C_sel] = xc @ [Bpw;Cpw]^T   (N=32, no bias)
  gemm(xc_bf, w_bc_bf, nullptr, bc, nullptr, 0, 32, 2);
  // 5) sequential scan
  ssm_scan<<<dim3(D_MODEL / 16, BATCH), 256, 0, stream>>>(delta, bc, xc, A_log, y);
  // 6) gate
  gate_mul<<<((int)MD + 255) / 256, 256, 0, stream>>>(y, xc, z, D_p, ybf, (int)MD);
  // 7) out = y @ out_w^T + out_b
  gemm(ybf, w_out_bf, outb, out, nullptr, 0, D_MODEL, 0);
}
